// MultiConvedAttention_43138651521189
// MI455X (gfx1250) — matrix operand order repaired and checked by reading, not compiled, not run on hardware
//
#include <hip/hip_runtime.h>

// ---------------------------------------------------------------------------
// MultiConvedAttention for MI455X (gfx1250): bf16 WMMA GEMMs + f32 accumulate
// GEMM1+GLU fused; 16x128 tile per wave (8 WMMA per A fragment), K=512 fixed.
// ---------------------------------------------------------------------------

typedef __attribute__((ext_vector_type(16))) __bf16 v16bf;
typedef __attribute__((ext_vector_type(8)))  __bf16 v8bf;
typedef __attribute__((ext_vector_type(8)))  float  v8f;

#define M_TOT   8192          // B*T = 16*512
#define C_FEAT  512
#define T_SEQ   512
#define N_HEAD  8
#define D_HEAD  64
#define GEMM_K  512

__device__ inline v16bf load_frag(const __bf16* lo, const __bf16* hi) {
  union { v16bf v; v8bf h[2]; } u;
  u.h[0] = *(const v8bf*)lo;
  u.h[1] = *(const v8bf*)hi;
  return u.v;
}

// ------------------------------- helpers -----------------------------------

__global__ void cvt_f32_bf16_kernel(const float* __restrict__ x,
                                    __bf16* __restrict__ y, int n) {
  int i = blockIdx.x * blockDim.x + threadIdx.x;
  if (i < n) y[i] = (__bf16)x[i];
}

// W (K x N, f32, row-major) -> Wt (N x K, bf16, row-major)
__global__ void prep_wt_kernel(const float* __restrict__ W,
                               __bf16* __restrict__ Wt, int K, int N) {
  int i = blockIdx.x * blockDim.x + threadIdx.x;
  if (i >= K * N) return;
  int k = i / N, n = i - k * N;
  Wt[(size_t)n * K + k] = (__bf16)W[i];
}

// w = softmax(h @ ww + wb) per row; ksz <= 7. One 64-thread block per row.
__global__ void rowdot_softmax_kernel(const float* __restrict__ h,
                                      const float* __restrict__ ww,
                                      const float* __restrict__ wb,
                                      float* __restrict__ wout, int ksz) {
  int row = blockIdx.x;
  int tid = threadIdx.x;
  __shared__ float part[7][64];
  float p[7];
#pragma unroll
  for (int j = 0; j < 7; ++j) p[j] = 0.f;
  const float* hr = h + (size_t)row * C_FEAT;
  for (int c = tid; c < C_FEAT; c += 64) {
    float hv = hr[c];
    for (int j = 0; j < ksz; ++j) p[j] += hv * ww[c * ksz + j];
  }
  for (int j = 0; j < ksz; ++j) part[j][tid] = p[j];
  __syncthreads();
  if (tid == 0) {
    float lg[7];
    float mx = -3.4028235e38f;
    for (int j = 0; j < ksz; ++j) {
      float s = wb[j];
      for (int l = 0; l < 64; ++l) s += part[j][l];
      lg[j] = s;
      mx = fmaxf(mx, s);
    }
    float sum = 0.f;
    for (int j = 0; j < ksz; ++j) { lg[j] = __expf(lg[j] - mx); sum += lg[j]; }
    float inv = 1.f / sum;
    for (int j = 0; j < ksz; ++j) wout[(size_t)row * ksz + j] = lg[j] * inv;
  }
}

// y[b,t,c] = sum_j w[b,t,j] * h[b, t+j-pad, c]; output bf16 (A of GEMM2)
__global__ void dynconv_kernel(const float* __restrict__ h,
                               const float* __restrict__ wc,
                               __bf16* __restrict__ y, int ksz, int pad) {
  int i = blockIdx.x * blockDim.x + threadIdx.x;
  if (i >= M_TOT * C_FEAT) return;
  int c = i & 511, r = i >> 9;
  int t = r & 511, b = r >> 9;
  const float* wr = wc + (size_t)r * ksz;
  float acc = 0.f;
  for (int j = 0; j < ksz; ++j) {
    int tt = t + j - pad;
    if (tt >= 0 && tt < T_SEQ)
      acc += wr[j] * h[((size_t)(b << 9) + tt) * C_FEAT + c];
  }
  y[i] = (__bf16)acc;
}

// ------------------------------ WMMA GEMM -----------------------------------
// D = A(MxK bf16) * Bt^T(Bt is NxK bf16) + bias. One wave -> 16x128 tile
// (8 WMMA tiles sharing one A fragment). K = 512 (compile-time).
// MODE 0: f32 out (ldo) + bias                      (grid.x = N/128)
// MODE 1: f32 out + bias + resid                    (grid.x = N/128)
// MODE 2: bf16 out scattered to (B, H, T, d)        (grid.x = N/128)
// MODE 3: bf16 out scattered to (B, H, d, T)        (grid.x = N/128)
// MODE 4: fused GLU: tiles 0-3 = value cols [nb,nb+64), tiles 4-7 = gate
//         cols [nb+512, nb+576); out h = a*sigmoid(g) -> f32 (outv) + bf16
//         (outv2).                                  (grid.x = 512/64 = 8)

template <int MODE>
__global__ __launch_bounds__(32)
void wmma_gemm_kernel(const __bf16* __restrict__ A,
                      const __bf16* __restrict__ Bt,
                      const float* __restrict__ bias,
                      const float* __restrict__ resid,
                      void* __restrict__ outv, void* __restrict__ outv2,
                      int ldo) {
  constexpr int K = GEMM_K;
  const int nb   = blockIdx.x * ((MODE == 4) ? 64 : 128);
  const int mb   = blockIdx.y * 16;
  const int lane = threadIdx.x;
  const int arow = mb + (lane & 15);
  const int akg  = (lane >> 4) * 16;   // A k-half     (contiguous, matches B)
  const int bkg  = (lane >> 4) * 16;   // B k-half     (contiguous, matches A)

  v8f acc[8] = {};
  const __bf16* arp = A + (size_t)arow * K;
  for (int k0 = 0; k0 < K; k0 += 32) {
    v16bf af = load_frag(arp + k0 + akg, arp + k0 + akg + 8);
#pragma unroll
    for (int tt = 0; tt < 8; ++tt) {
      int col;
      if constexpr (MODE == 4)
        col = (tt < 4) ? (nb + tt * 16) : (512 + nb + (tt - 4) * 16);
      else
        col = nb + tt * 16;
      const __bf16* bp =
          Bt + (size_t)(col + (lane & 15)) * K + k0 + bkg;
      v16bf bf = load_frag(bp, bp + 8);
      acc[tt] = __builtin_amdgcn_wmma_f32_16x16x32_bf16(
          false, af, false, bf, (short)0, acc[tt], false, false);
    }
  }

  if constexpr (MODE == 4) {
    // fused GLU epilogue
#pragma unroll
    for (int tt = 0; tt < 4; ++tt) {
#pragma unroll
      for (int i = 0; i < 8; ++i) {
        int row = mb + i + ((lane >> 4) << 3);
        int col = nb + tt * 16 + (lane & 15);
        float a = acc[tt][i]     + bias[col];
        float g = acc[tt + 4][i] + bias[col + 512];
        float v = a * (1.f / (1.f + __expf(-g)));
        ((float*)outv)[(size_t)row * C_FEAT + col]   = v;
        ((__bf16*)outv2)[(size_t)row * C_FEAT + col] = (__bf16)v;
      }
    }
  } else {
#pragma unroll
    for (int tt = 0; tt < 8; ++tt) {
#pragma unroll
      for (int i = 0; i < 8; ++i) {
        int row = mb + i + ((lane >> 4) << 3);
        int col = nb + tt * 16 + (lane & 15);
        float v = acc[tt][i] + bias[col];
        if constexpr (MODE == 0) {
          ((float*)outv)[(size_t)row * ldo + col] = v;
        } else if constexpr (MODE == 1) {
          ((float*)outv)[(size_t)row * ldo + col] =
              v + resid[(size_t)row * ldo + col];
        } else if constexpr (MODE == 2) {
          int b = row >> 9, t = row & 511, hh = col >> 6, d = col & 63;
          ((__bf16*)outv)[((((size_t)b * N_HEAD + hh) * T_SEQ) + t) * D_HEAD + d] =
              (__bf16)v;
        } else if constexpr (MODE == 3) {
          int b = row >> 9, t = row & 511, hh = col >> 6, d = col & 63;
          ((__bf16*)outv)[((((size_t)b * N_HEAD + hh) * D_HEAD) + d) * T_SEQ + t] =
              (__bf16)v;
        }
      }
    }
  }
}

// ------------------------------ attention -----------------------------------
// grid (T/16, H, B), 256 threads = 8 waves. Wave w owns key range [w*64,w*64+64).
__global__ __launch_bounds__(256)
void attention_kernel(const __bf16* __restrict__ qp,   // (B,H,T,d)
                      const __bf16* __restrict__ kp,   // (B,H,T,d)
                      const __bf16* __restrict__ vT,   // (B,H,d,T)
                      const int* __restrict__ mask,    // (B,1,T)
                      __bf16* __restrict__ xout) {     // (B,T,C) bf16
  const int qt0  = blockIdx.x * 16;
  const int h    = blockIdx.y;
  const int b    = blockIdx.z;
  const int tid  = threadIdx.x;
  const int wave = tid >> 5;
  const int lane = tid & 31;

  __shared__ float  sS[16][T_SEQ];        // 32 KB scores
  __shared__ __bf16 sP[16][T_SEQ];        // 16 KB probs (bf16 for WMMA A)
  __shared__ float  sO[16][D_HEAD];       // 4 KB partial outputs
  __shared__ float  sRed[16][16];
  __shared__ float  sMax[16], sSum[16];

  for (int i = tid; i < 16 * D_HEAD; i += 256) ((float*)sO)[i] = 0.f;

  const __bf16* qbase = qp + (((size_t)b * N_HEAD + h) * T_SEQ + qt0) * D_HEAD;
  const __bf16* kbase = kp + (((size_t)b * N_HEAD + h) * T_SEQ) * D_HEAD;
  const __bf16* vbase = vT + (((size_t)b * N_HEAD + h) * D_HEAD) * T_SEQ;
  const int key0 = wave * 64;
  const int akg = (lane >> 4) * 16;
  const int bkg = (lane >> 4) * 16;
  const int arow = lane & 15;

  // ---- scores: Q (16 x 64) @ K^T (64 x 64 keys) ----
  v8f acc[4] = {};
  for (int kk = 0; kk < 2; ++kk) {
    const __bf16* ap = qbase + (size_t)arow * D_HEAD + kk * 32 + akg;
    v16bf af = load_frag(ap, ap + 8);
#pragma unroll
    for (int tt = 0; tt < 4; ++tt) {
      int kc = key0 + tt * 16 + (lane & 15);
      const __bf16* bp = kbase + (size_t)kc * D_HEAD + kk * 32 + bkg;
      v16bf bf = load_frag(bp, bp + 8);
      acc[tt] = __builtin_amdgcn_wmma_f32_16x16x32_bf16(
          false, af, false, bf, (short)0, acc[tt], false, false);
    }
  }
#pragma unroll
  for (int tt = 0; tt < 4; ++tt) {
#pragma unroll
    for (int i = 0; i < 8; ++i) {
      int r = i + ((lane >> 4) << 3);
      int c = key0 + tt * 16 + (lane & 15);
      float s = acc[tt][i] * 0.125f;                 // 1/sqrt(64)
      if (mask[b * T_SEQ + c] == 0) s = -3.4028235e38f;
      sS[r][c] = s;
    }
  }
  __syncthreads();

  // ---- softmax over keys (16 threads per row) ----
  {
    int r = tid >> 4, cl = tid & 15;
    float mx = -3.4028235e38f;
    for (int c = cl; c < T_SEQ; c += 16) mx = fmaxf(mx, sS[r][c]);
    sRed[r][cl] = mx;
    __syncthreads();
    if (cl == 0) {
      float m = sRed[r][0];
      for (int l = 1; l < 16; ++l) m = fmaxf(m, sRed[r][l]);
      sMax[r] = m;
    }
    __syncthreads();
    float m = sMax[r];
    float sum = 0.f;
    for (int c = cl; c < T_SEQ; c += 16) {
      float e = __expf(sS[r][c] - m);
      sS[r][c] = e;
      sum += e;
    }
    sRed[r][cl] = sum;
    __syncthreads();
    if (cl == 0) {
      float s = 0.f;
      for (int l = 0; l < 16; ++l) s += sRed[r][l];
      sSum[r] = s;
    }
    __syncthreads();
    float inv = 1.f / sSum[r];
    for (int c = cl; c < T_SEQ; c += 16) {
      float p = sS[r][c] * inv;
      if (mask[b * T_SEQ + c] == 0) p = 0.f;        // second masked_fill
      sP[r][c] = (__bf16)p;
    }
  }
  __syncthreads();

  // ---- P (16 x 64 keys) @ V (64 keys x 64 d), reduce across waves ----
  v8f oacc[4] = {};
  for (int kk = 0; kk < 2; ++kk) {
    int kb = key0 + kk * 32;
    const __bf16* ap = &sP[arow][kb + akg];
    v16bf af = load_frag(ap, ap + 8);
#pragma unroll
    for (int tt = 0; tt < 4; ++tt) {
      int d = tt * 16 + (lane & 15);
      const __bf16* bp = vbase + (size_t)d * T_SEQ + kb + bkg;
      v16bf bf = load_frag(bp, bp + 8);
      oacc[tt] = __builtin_amdgcn_wmma_f32_16x16x32_bf16(
          false, af, false, bf, (short)0, oacc[tt], false, false);
    }
  }
#pragma unroll
  for (int tt = 0; tt < 4; ++tt) {
#pragma unroll
    for (int i = 0; i < 8; ++i) {
      int r = i + ((lane >> 4) << 3);
      int d = tt * 16 + (lane & 15);
      atomicAdd(&sO[r][d], oacc[tt][i]);
    }
  }
  __syncthreads();

  for (int i = tid; i < 16 * D_HEAD; i += 256) {
    int r = i >> 6, d = i & 63;
    int trow = qt0 + r;
    xout[((size_t)(b * T_SEQ + trow)) * C_FEAT + h * D_HEAD + d] =
        (__bf16)sO[r][d];
  }
}

// ------------------------------- launch -------------------------------------

extern "C" void kernel_launch(void* const* d_in, const int* in_sizes, int n_in,
                              void* d_out, int out_size, void* d_ws,
                              size_t ws_size, hipStream_t stream) {
  (void)in_sizes; (void)n_in; (void)out_size; (void)ws_size;
  const float* query = (const float*)d_in[0];
  const float* key   = (const float*)d_in[1];
  const float* value = (const float*)d_in[2];
  const int*   mask  = (const int*)d_in[3];

  const int M = M_TOT, n = M_TOT * C_FEAT;

  char* wsp = (char*)d_ws;
  auto alloc = [&](size_t bytes) -> void* {
    void* p = (void*)wsp;
    wsp += (bytes + 255) & ~(size_t)255;
    return p;
  };

  __bf16* qbf    = (__bf16*)alloc((size_t)n * 2);
  __bf16* kbf    = (__bf16*)alloc((size_t)n * 2);
  __bf16* vbf    = (__bf16*)alloc((size_t)n * 2);
  float*  kconv  = (float*) alloc((size_t)n * 4);
  float*  vconv  = (float*) alloc((size_t)n * 4);
  float*  hbuff  = (float*) alloc((size_t)n * 4);
  __bf16* hbufb  = (__bf16*)alloc((size_t)n * 2);
  float*  wcoef  = (float*) alloc((size_t)M * 7 * 4);
  __bf16* ybf    = (__bf16*)alloc((size_t)n * 2);
  __bf16* w1t    = (__bf16*)alloc((size_t)1024 * 512 * 2);
  __bf16* w2t    = (__bf16*)alloc((size_t)128 * 512 * 2);
  __bf16* wt     = (__bf16*)alloc((size_t)512 * 512 * 2);
  __bf16* kcbf   = (__bf16*)alloc((size_t)n * 2);
  __bf16* vcbf   = (__bf16*)alloc((size_t)n * 2);
  __bf16* qproj  = (__bf16*)alloc((size_t)n * 2);
  __bf16* kproj  = (__bf16*)alloc((size_t)n * 2);
  __bf16* vprojT = (__bf16*)alloc((size_t)n * 2);
  __bf16* xbuf   = (__bf16*)alloc((size_t)n * 2);

  const int TB = 256;
  const int NB = (n + TB - 1) / TB;

  cvt_f32_bf16_kernel<<<NB, TB, 0, stream>>>(query, qbf, n);
  cvt_f32_bf16_kernel<<<NB, TB, 0, stream>>>(key,   kbf, n);
  cvt_f32_bf16_kernel<<<NB, TB, 0, stream>>>(value, vbf, n);

  static const int KSZ[4] = {1, 3, 5, 7};
  for (int s = 0; s < 8; ++s) {
    int br = s & 3;
    bool isK = s < 4;
    int base = (isK ? 4 : 28) + br * 6;
    const float* w1 = (const float*)d_in[base + 0];
    const float* b1 = (const float*)d_in[base + 1];
    const float* ww = (const float*)d_in[base + 2];
    const float* wb = (const float*)d_in[base + 3];
    const float* w2 = (const float*)d_in[base + 4];
    const float* b2 = (const float*)d_in[base + 5];
    const __bf16* srcb = isK ? kbf : vbf;
    const float*  srcf = isK ? key : value;
    float* outc = isK ? kconv : vconv;
    int coloff = br * 128;
    int ks = KSZ[br], pad = (ks - 1) / 2;

    prep_wt_kernel<<<(512 * 1024 + TB - 1) / TB, TB, 0, stream>>>(w1, w1t, 512, 1024);
    // fused GEMM1 + GLU: 8 x 512 blocks, each wave does 4 value + 4 gate tiles
    wmma_gemm_kernel<4><<<dim3(8, M / 16), 32, 0, stream>>>(
        srcb, w1t, b1, nullptr, hbuff, hbufb, C_FEAT);
    rowdot_softmax_kernel<<<M, 64, 0, stream>>>(hbuff, ww, wb, wcoef, ks);
    dynconv_kernel<<<NB, TB, 0, stream>>>(hbuff, wcoef, ybf, ks, pad);
    prep_wt_kernel<<<(512 * 128 + TB - 1) / TB, TB, 0, stream>>>(w2, w2t, 512, 128);
    wmma_gemm_kernel<1><<<dim3(128 / 128, M / 16), 32, 0, stream>>>(
        ybf, w2t, b2, srcf + coloff, outc + coloff, nullptr, 512);
  }

  cvt_f32_bf16_kernel<<<NB, TB, 0, stream>>>(kconv, kcbf, n);
  cvt_f32_bf16_kernel<<<NB, TB, 0, stream>>>(vconv, vcbf, n);

  const float* wq = (const float*)d_in[52];
  const float* wk = (const float*)d_in[53];
  const float* wv = (const float*)d_in[54];
  const float* wo = (const float*)d_in[55];
  const float* bq = (const float*)d_in[56];
  const float* bk = (const float*)d_in[57];
  const float* bv = (const float*)d_in[58];
  const float* bo = (const float*)d_in[59];

  const int WN = (512 * 512 + TB - 1) / TB;
  prep_wt_kernel<<<WN, TB, 0, stream>>>(wq, wt, 512, 512);
  wmma_gemm_kernel<2><<<dim3(512 / 128, M / 16), 32, 0, stream>>>(
      qbf, wt, bq, nullptr, qproj, nullptr, 512);
  prep_wt_kernel<<<WN, TB, 0, stream>>>(wk, wt, 512, 512);
  wmma_gemm_kernel<2><<<dim3(512 / 128, M / 16), 32, 0, stream>>>(
      kcbf, wt, bk, nullptr, kproj, nullptr, 512);
  prep_wt_kernel<<<WN, TB, 0, stream>>>(wv, wt, 512, 512);
  wmma_gemm_kernel<3><<<dim3(512 / 128, M / 16), 32, 0, stream>>>(
      vcbf, wt, bv, nullptr, vprojT, nullptr, 512);

  attention_kernel<<<dim3(T_SEQ / 16, N_HEAD, 16), 256, 0, stream>>>(
      qproj, kproj, vprojT, mask, xbuf);

  prep_wt_kernel<<<WN, TB, 0, stream>>>(wo, wt, 512, 512);
  wmma_gemm_kernel<0><<<dim3(512 / 128, M / 16), 32, 0, stream>>>(
      xbuf, wt, bo, nullptr, (float*)d_out, nullptr, 512);
}
